// NonCrossAttention_17746804867816
// MI455X (gfx1250) — compile-verified, matrix operand order checked by reading, not run on hardware
//
#include <hip/hip_runtime.h>
#include <hip/hip_bf16.h>

// Problem constants
#define B_   8
#define C_   256
#define CI_  128
#define N_   4096   // H*W
#define M_   1024   // (H/2)*(W/2)
#define PSTR 1044   // padded LDS row stride (bf16 elems) for P tiles: 522 words % 64 = 10 -> conflict-free
#define WSTR 264    // padded LDS row stride (bf16 elems) for staged weights: 132 words % 64 = 4 -> conflict-free

typedef __attribute__((ext_vector_type(16))) __bf16 v16bf;
typedef __attribute__((ext_vector_type(8)))  float  v8f;

union TileU  { v16bf v; unsigned long long q[4]; };
union Bf16U  { __bf16 h; unsigned short u; };

static __device__ inline v8f wmma_bf16(v16bf a, v16bf b, v8f c) {
  return __builtin_amdgcn_wmma_f32_16x16x32_bf16(false, a, false, b, (short)0, c, false, false);
}

// A/B fragment load, K-dimension contiguous in memory.
// rowbase = pointer to this lane's row (A: row = lane&15) or col (B: col = lane&15).
// Per ISA layout: VGPR v in 32-K group g holds K = base + half*8 + 2*(v&3) + {0,1}.
static __device__ inline v16bf load_tile_g(const __bf16* rowbase, int k0, int hf) {
  TileU t;
  const __bf16* p = rowbase + k0 + hf * 8;
  t.q[0] = *(const unsigned long long*)(p);
  t.q[1] = *(const unsigned long long*)(p + 4);
  t.q[2] = *(const unsigned long long*)(p + 16);
  t.q[3] = *(const unsigned long long*)(p + 20);
  return t.v;
}
static __device__ inline v16bf load_tile_lds(const unsigned short* rowbase, int k0, int hf) {
  TileU t;
  const unsigned short* p = rowbase + k0 + hf * 8;
  t.q[0] = *(const unsigned long long*)(p);
  t.q[1] = *(const unsigned long long*)(p + 4);
  t.q[2] = *(const unsigned long long*)(p + 16);
  t.q[3] = *(const unsigned long long*)(p + 20);
  return t.v;
}

// Async global->LDS copy of 16 bytes per lane (ASYNCcnt-tracked, CDNA5).
// lds_base is passed as an operand (even though unused in the template) so the
// shared object ESCAPES into the asm: combined with the "memory" clobber this
// forces LLVM to treat the asm as a writer of the LDS array (otherwise capture
// analysis proves the array is never written and folds later reads to undef).
static __device__ inline void async_g2l_b128(unsigned lds_byte_off, const void* gaddr,
                                             const void* lds_base) {
  asm volatile("global_load_async_to_lds_b128 %0, %1, off"
               :: "v"(lds_byte_off), "v"(gaddr), "v"(lds_base)
               : "memory");
}
static __device__ inline void wait_async0() {
  asm volatile("s_wait_asynccnt 0" ::: "memory");
}

// ---------------- prep kernels ----------------

// [B][C][N] f32  ->  [B][N][C] bf16  (LDS tiled transpose, coalesced both sides)
__global__ __launch_bounds__(256) void k_transpose_bf16(const float* __restrict__ src,
                                                        __bf16* __restrict__ dst) {
  __shared__ float tile[32][33];
  int n0 = blockIdx.x * 32, c0 = blockIdx.y * 32;
  const float* s = src + (size_t)blockIdx.z * C_ * N_;
  __bf16*      d = dst + (size_t)blockIdx.z * C_ * N_;
  int tx = threadIdx.x, ty = threadIdx.y;
#pragma unroll
  for (int j = 0; j < 4; ++j)
    tile[ty + 8*j][tx] = s[(size_t)(c0 + ty + 8*j) * N_ + n0 + tx];
  __syncthreads();
#pragma unroll
  for (int j = 0; j < 4; ++j)
    d[(size_t)(n0 + ty + 8*j) * C_ + c0 + tx] = (__bf16)tile[tx][ty + 8*j];
}

__global__ void k_f32_to_bf16(const float* __restrict__ s, __bf16* __restrict__ d, int n) {
  int i = blockIdx.x * 256 + threadIdx.x;
  if (i < n) d[i] = (__bf16)s[i];
}

// ---------------- conv1x1 as GEMM: Y[n,co] = X[n,:] @ Wt[co,:]^T + bias[co] ----------------
// X: [B*N][256] bf16, Wt: [128][256] bf16, Y: [B*N][128] bf16. 4 waves/block, 16 rows/wave.
// Weights (64 KB) staged once per block into LDS via async-to-LDS, padded rows (bank-skewed).
__global__ __launch_bounds__(128) void k_gemm_nc(const __bf16* __restrict__ X,
                                                 const __bf16* __restrict__ Wt,
                                                 const float* __restrict__ bias,
                                                 __bf16* __restrict__ Y) {
  __shared__ __align__(16) __bf16 wlds[CI_ * WSTR];   // only LDS object in this kernel -> offset 0
  int tid = threadIdx.x;

  // stage Wt[128][256] -> wlds (16B chunks; 4096 chunks over 128 threads)
  for (int c = tid; c < CI_ * 32; c += 128) {
    int row = c >> 5, j = c & 31;                      // j: 8-element (16B) group within row
    unsigned lds_off = (unsigned)((row * WSTR + j * 8) * 2);
    async_g2l_b128(lds_off, (const void*)(Wt + row * C_ + j * 8), (const void*)wlds);
  }
  wait_async0();
  __syncthreads();

  int wv = tid >> 5, lane = tid & 31, hf = lane >> 4, l15 = lane & 15;
  size_t n0 = (size_t)blockIdx.x * 64 + (size_t)wv * 16;
  const __bf16* Xrow = X + (n0 + l15) * C_;
  v16bf a[8];
#pragma unroll
  for (int k = 0; k < 8; ++k) a[k] = load_tile_g(Xrow, 32*k, hf);

#pragma unroll
  for (int ct = 0; ct < 8; ++ct) {
    const unsigned short* Wrow =
        (const unsigned short*)wlds + (size_t)(ct*16 + l15) * WSTR;
    v16bf bw[8];
#pragma unroll
    for (int k = 0; k < 8; ++k) bw[k] = load_tile_lds(Wrow, 32*k, hf);
    float bs = bias[ct*16 + l15];       // D column = lane&15 -> per-co bias
    v8f acc;
#pragma unroll
    for (int i = 0; i < 8; ++i) acc[i] = bs;
#pragma unroll
    for (int k = 0; k < 8; ++k) acc = wmma_bf16(a[k], bw[k], acc);
#pragma unroll
    for (int v = 0; v < 8; ++v)
      Y[(n0 + v + 8*hf) * CI_ + ct*16 + l15] = (__bf16)acc[v];
  }
}

// ---------------- 2x2 maxpool: full-res [B][N][Ci] -> K [B][M][Ci] and V [B][Ci][M] ----------------
__global__ void k_pool_K(const __bf16* __restrict__ P, __bf16* __restrict__ K) {
  int i = blockIdx.x * 256 + threadIdx.x;           // B*M*CI threads
  int ci = i & (CI_-1); int m = (i >> 7) & (M_-1); int b = i >> 17;
  int ph = m >> 5, pw = m & 31;
  size_t base = ((size_t)b * N_ + ph*128 + pw*2) * CI_ + ci;
  float v0 = (float)P[base],          v1 = (float)P[base + CI_];
  float v2 = (float)P[base + 64*CI_], v3 = (float)P[base + 65*CI_];
  K[i] = (__bf16)fmaxf(fmaxf(v0, v1), fmaxf(v2, v3));
}
__global__ void k_pool_V(const __bf16* __restrict__ G, __bf16* __restrict__ V) {
  int i = blockIdx.x * 256 + threadIdx.x;
  int m = i & (M_-1); int ci = (i >> 10) & (CI_-1); int b = i >> 17;
  int ph = m >> 5, pw = m & 31;
  size_t base = ((size_t)b * N_ + ph*128 + pw*2) * CI_ + ci;
  float v0 = (float)G[base],          v1 = (float)G[base + CI_];
  float v2 = (float)G[base + 64*CI_], v3 = (float)G[base + 65*CI_];
  V[i] = (__bf16)fmaxf(fmaxf(v0, v1), fmaxf(v2, v3));
}

// ---------------- attention: softmax(Q Kᵀ) V for a 16-row query tile ----------------
// 8 waves; wave w owns keys [128w,128w+128) for logits, then ci-tile w for the PV GEMM.
__global__ __launch_bounds__(256) void k_attention(const __bf16* __restrict__ Q,
                                                   const __bf16* __restrict__ K,
                                                   const __bf16* __restrict__ V,
                                                   __bf16* __restrict__ O) {
  __shared__ unsigned short p_lds[16 * PSTR];   // exp'd probs, bf16, bank-skewed rows
  __shared__ float red_lds[8 * 16];
  __shared__ float rowmax_s[16];
  __shared__ float rowsum_s[16];

  int tid = threadIdx.x;
  int wv = tid >> 5, lane = tid & 31, hf = lane >> 4, l15 = lane & 15;
  int b  = blockIdx.x >> 8;
  int n0 = (blockIdx.x & 255) << 4;

  // Q fragments for 16 rows x 128 ci (reused across all key tiles)
  const __bf16* Qrow = Q + ((size_t)b * N_ + n0 + l15) * CI_;
  v16bf a0 = load_tile_g(Qrow,  0, hf);
  v16bf a1 = load_tile_g(Qrow, 32, hf);
  v16bf a2 = load_tile_g(Qrow, 64, hf);
  v16bf a3 = load_tile_g(Qrow, 96, hf);

  int m_base = wv << 7;
  const __bf16* Kbase = K + ((size_t)b * M_ + m_base + l15) * CI_;

  // double-buffered K tiles: issue next tile's loads before current WMMAs
  v16bf bA0 = load_tile_g(Kbase,  0, hf);
  v16bf bA1 = load_tile_g(Kbase, 32, hf);
  v16bf bA2 = load_tile_g(Kbase, 64, hf);
  v16bf bA3 = load_tile_g(Kbase, 96, hf);

  v8f f[8];
#pragma unroll
  for (int mt = 0; mt < 8; ++mt) {
    v16bf bN0, bN1, bN2, bN3;
    if (mt < 7) {
      const __bf16* Kr = Kbase + (size_t)(mt + 1) * 16 * CI_;
      __builtin_prefetch(Kr + 16*CI_, 0, 0);
      bN0 = load_tile_g(Kr,  0, hf);
      bN1 = load_tile_g(Kr, 32, hf);
      bN2 = load_tile_g(Kr, 64, hf);
      bN3 = load_tile_g(Kr, 96, hf);
    }
    v8f acc = {};
    acc = wmma_bf16(a0, bA0, acc);
    acc = wmma_bf16(a1, bA1, acc);
    acc = wmma_bf16(a2, bA2, acc);
    acc = wmma_bf16(a3, bA3, acc);
    f[mt] = acc;
    if (mt < 7) { bA0 = bN0; bA1 = bN1; bA2 = bN2; bA3 = bN3; }
  }

  // ---- row max: per-lane over 8 key tiles, across the 16-lane half, then across waves
#pragma unroll
  for (int v = 0; v < 8; ++v) {
    float pm = f[0][v];
#pragma unroll
    for (int mt = 1; mt < 8; ++mt) pm = fmaxf(pm, f[mt][v]);
#pragma unroll
    for (int off = 1; off < 16; off <<= 1) pm = fmaxf(pm, __shfl_xor(pm, off, 16));
    if (l15 == 0) red_lds[wv * 16 + 8*hf + v] = pm;
  }
  __syncthreads();
  if (tid < 16) {
    float mx = red_lds[tid];
    for (int w = 1; w < 8; ++w) mx = fmaxf(mx, red_lds[w*16 + tid]);
    rowmax_s[tid] = mx;
  }
  __syncthreads();

  // ---- exp + row sum; write bf16 probabilities to LDS (unnormalized)
#pragma unroll
  for (int v = 0; v < 8; ++v) {
    float rm = rowmax_s[8*hf + v];
    float s = 0.f;
#pragma unroll
    for (int mt = 0; mt < 8; ++mt) {
      float e = __expf(f[mt][v] - rm);
      s += e;
      Bf16U u; u.h = (__bf16)e;
      p_lds[(8*hf + v) * PSTR + m_base + mt*16 + l15] = u.u;
    }
#pragma unroll
    for (int off = 1; off < 16; off <<= 1) s += __shfl_xor(s, off, 16);
    if (l15 == 0) red_lds[wv * 16 + 8*hf + v] = s;
  }
  __syncthreads();
  if (tid < 16) {
    float s = 0.f;
    for (int w = 0; w < 8; ++w) s += red_lds[w*16 + tid];
    rowsum_s[tid] = s;
  }
  __syncthreads();

  // ---- PV GEMM: wave w computes out[16, 16] for ci-tile w, K = 1024 keys (double-buffered V)
  int ci0 = wv << 4;
  const unsigned short* Prow = p_lds + (size_t)l15 * PSTR;        // A row = query row
  const __bf16* Vrow = V + ((size_t)b * CI_ + ci0 + l15) * M_;    // B col = ci, K = m contiguous
  v8f oacc = {};
  v16bf vcur = load_tile_g(Vrow, 0, hf);
#pragma unroll 4
  for (int mc = 0; mc < 32; ++mc) {
    v16bf vnext;
    if (mc < 31) vnext = load_tile_g(Vrow, (mc + 1) * 32, hf);
    v16bf pa = load_tile_lds(Prow, mc * 32, hf);
    oacc = wmma_bf16(pa, vcur, oacc);
    if (mc < 31) vcur = vnext;
  }
#pragma unroll
  for (int v = 0; v < 8; ++v) {
    int row = v + 8*hf;
    float o = oacc[v] * (1.0f / rowsum_s[row]);   // fold softmax normalization
    O[((size_t)b * N_ + n0 + row) * CI_ + ci0 + l15] = (__bf16)o;
  }
}

// ---------------- W projection: WY[b,c,n] = sum_ci Ww[c,ci]*A[b,n,ci] + wb[c] ----------------
// Rows = c so fp32 stores into [B][C][N] are lane-coalesced.
__global__ __launch_bounds__(128) void k_wproj(const __bf16* __restrict__ A,
                                               const __bf16* __restrict__ Wt,
                                               const float* __restrict__ bias,
                                               float* __restrict__ WY) {
  int tid = threadIdx.x;
  int wv = tid >> 5, lane = tid & 31, hf = lane >> 4, l15 = lane & 15;
  int bid = blockIdx.x;
  int b   = bid >> 8;
  int nch = (bid >> 2) & 63;
  int cch = bid & 3;
  int c0  = cch * 64 + wv * 16;
  const __bf16* Wrow = Wt + (size_t)(c0 + l15) * CI_;
  v16bf a[4];
#pragma unroll
  for (int k = 0; k < 4; ++k) a[k] = load_tile_g(Wrow, 32*k, hf);
#pragma unroll
  for (int nt = 0; nt < 4; ++nt) {
    int n0 = nch * 64 + nt * 16;
    const __bf16* Brow = A + ((size_t)b * N_ + n0 + l15) * CI_;
    if (nt < 3) __builtin_prefetch(Brow + 16*CI_, 0, 0);
    v16bf bb[4];
#pragma unroll
    for (int k = 0; k < 4; ++k) bb[k] = load_tile_g(Brow, 32*k, hf);
    v8f acc;
#pragma unroll
    for (int v = 0; v < 8; ++v) acc[v] = bias[c0 + v + 8*hf];   // per-row (c) bias
#pragma unroll
    for (int k = 0; k < 4; ++k) acc = wmma_bf16(a[k], bb[k], acc);
#pragma unroll
    for (int v = 0; v < 8; ++v)
      WY[(size_t)b * C_ * N_ + (size_t)(c0 + v + 8*hf) * N_ + n0 + l15] = acc[v];
  }
}

// ---------------- BatchNorm (training stats) ----------------
__global__ __launch_bounds__(256) void k_bn_stats(const float* __restrict__ WY,
                                                  const float* __restrict__ gamma,
                                                  const float* __restrict__ beta,
                                                  float* __restrict__ scale,
                                                  float* __restrict__ shift) {
  int c = blockIdx.x, t = threadIdx.x;
  float s = 0.f, s2 = 0.f;
  for (int b = 0; b < B_; ++b) {
    const float* p = WY + ((size_t)b * C_ + c) * N_;
    for (int j = t; j < N_; j += 256) { float v = p[j]; s += v; s2 += v*v; }
  }
  __shared__ float rs[256], rs2[256];
  rs[t] = s; rs2[t] = s2; __syncthreads();
  for (int o = 128; o > 0; o >>= 1) {
    if (t < o) { rs[t] += rs[t+o]; rs2[t] += rs2[t+o]; }
    __syncthreads();
  }
  if (t == 0) {
    float inv_n = 1.0f / (float)(B_ * N_);
    float mean = rs[0] * inv_n;
    float var  = rs2[0] * inv_n - mean * mean;
    float r    = rsqrtf(var + 1e-5f);
    float sc   = gamma[c] * r;
    scale[c] = sc;
    shift[c] = beta[c] - mean * sc;
  }
}

__global__ void k_bn_apply(const float* __restrict__ WY, const float* __restrict__ x,
                           const float* __restrict__ scale, const float* __restrict__ shift,
                           float* __restrict__ out) {
  int i = blockIdx.x * 256 + threadIdx.x;          // B*C*N = 8388608
  int c = (i >> 12) & (C_ - 1);
  out[i] = WY[i] * scale[c] + shift[c] + x[i];
}

// ---------------- host orchestration ----------------
extern "C" void kernel_launch(void* const* d_in, const int* in_sizes, int n_in,
                              void* d_out, int out_size, void* d_ws, size_t ws_size,
                              hipStream_t stream) {
  const float* x       = (const float*)d_in[0];
  const float* y       = (const float*)d_in[1];
  const float* theta_w = (const float*)d_in[2];
  const float* theta_b = (const float*)d_in[3];
  const float* phi_w   = (const float*)d_in[4];
  const float* phi_b   = (const float*)d_in[5];
  const float* g_w     = (const float*)d_in[6];
  const float* g_b     = (const float*)d_in[7];
  const float* w_w     = (const float*)d_in[8];
  const float* w_b     = (const float*)d_in[9];
  const float* gamma   = (const float*)d_in[10];
  const float* beta    = (const float*)d_in[11];
  float* out = (float*)d_out;

  const size_t SZ_BNC  = (size_t)B_ * N_ * C_  * 2;   // 16 MB
  const size_t SZ_BNCI = (size_t)B_ * N_ * CI_ * 2;   // 8 MB
  const size_t SZ_BMCI = (size_t)B_ * M_ * CI_ * 2;   // 2 MB
  const size_t SZ_W    = (size_t)C_ * CI_ * 2;        // 64 KB
  const size_t SZ_WY   = (size_t)B_ * C_ * N_ * 4;    // 32 MB

  char* ws = (char*)d_ws;
  size_t off = 0;
  __bf16* xb   = (__bf16*)(ws + off); off += SZ_BNC;
  __bf16* yb   = (__bf16*)(ws + off); off += SZ_BNC;
  __bf16* qb   = (__bf16*)(ws + off); off += SZ_BNCI;
  __bf16* phif = (__bf16*)(ws + off); off += SZ_BNCI;
  __bf16* gf   = (__bf16*)(ws + off); off += SZ_BNCI;
  __bf16* ao   = (__bf16*)(ws + off); off += SZ_BNCI;
  __bf16* kb   = (__bf16*)(ws + off); off += SZ_BMCI;
  __bf16* vb   = (__bf16*)(ws + off); off += SZ_BMCI;
  float*  wy   = (float*) (ws + off); off += SZ_WY;
  __bf16* twb  = (__bf16*)(ws + off); off += SZ_W;
  __bf16* pwb  = (__bf16*)(ws + off); off += SZ_W;
  __bf16* gwb  = (__bf16*)(ws + off); off += SZ_W;
  __bf16* wwb  = (__bf16*)(ws + off); off += SZ_W;
  float*  sc   = (float*) (ws + off); off += 1024;
  float*  sh   = (float*) (ws + off); off += 1024;

  // 1) layout prep: NCHW f32 -> [N,C] bf16; weights -> bf16
  dim3 tgrid(N_/32, C_/32, B_), tblk(32, 8);
  k_transpose_bf16<<<tgrid, tblk, 0, stream>>>(x, xb);
  k_transpose_bf16<<<tgrid, tblk, 0, stream>>>(y, yb);
  int nw = C_ * CI_;
  k_f32_to_bf16<<<(nw + 255)/256, 256, 0, stream>>>(theta_w, twb, nw);
  k_f32_to_bf16<<<(nw + 255)/256, 256, 0, stream>>>(phi_w,   pwb, nw);
  k_f32_to_bf16<<<(nw + 255)/256, 256, 0, stream>>>(g_w,     gwb, nw);
  k_f32_to_bf16<<<(nw + 255)/256, 256, 0, stream>>>(w_w,     wwb, nw);

  // 2) 1x1 convs as WMMA GEMMs (weights async-staged to LDS)
  k_gemm_nc<<<(B_*N_)/64, 128, 0, stream>>>(xb, twb, theta_b, qb);
  k_gemm_nc<<<(B_*N_)/64, 128, 0, stream>>>(yb, pwb, phi_b,   phif);
  k_gemm_nc<<<(B_*N_)/64, 128, 0, stream>>>(yb, gwb, g_b,     gf);

  // 3) 2x2 maxpool into attention-friendly layouts
  k_pool_K<<<(B_*M_*CI_)/256, 256, 0, stream>>>(phif, kb);
  k_pool_V<<<(B_*M_*CI_)/256, 256, 0, stream>>>(gf,   vb);

  // 4) attention (QK^T, softmax, PV) — one block per 16 query rows
  k_attention<<<B_*(N_/16), 256, 0, stream>>>(qb, kb, vb, ao);

  // 5) W projection into [B,C,N] f32
  k_wproj<<<B_*(N_/64)*(C_/64), 128, 0, stream>>>(ao, wwb, w_b, wy);

  // 6) BatchNorm stats + apply + residual
  k_bn_stats<<<C_, 256, 0, stream>>>(wy, gamma, beta, sc, sh);
  k_bn_apply<<<(B_*C_*N_)/256, 256, 0, stream>>>(wy, x, sc, sh, out);
}
